// Model_Recursive_LSTM_v2_76690936038050
// MI455X (gfx1250) — compile-verified, matrix operand order checked by reading, not run on hardware
//
#include <hip/hip_runtime.h>
#include <hip/hip_bf16.h>

// ---------------------------------------------------------------------------
// Model_Recursive_LSTM_v2 forward pass for gfx1250 (MI455X, wave32, WMMA).
//
// All GEMM-shaped work (LSTM gate matmuls, MLP layers) runs through
// V_WMMA_F32_16X16X32_F16.  Weights are pre-swizzled each call into
// "fragment-major" f16 (16 contiguous halves per lane matching the CDNA5
// 16-bit A/B VGPR layout), so the hot loop is ds_load_b128 (A from LDS) +
// global_load_b128 (B from L2) + v_wmma.  LSTM recurrences are per-batch-row
// independent: each block owns 16 rows, keeps h/c in LDS, loops all T steps.
// ---------------------------------------------------------------------------

typedef _Float16 half_t;
typedef __attribute__((ext_vector_type(16))) _Float16 v16h;
typedef __attribute__((ext_vector_type(8)))  _Float16 v8h;
typedef __attribute__((ext_vector_type(8)))  float    v8f;

__device__ __forceinline__ float sigmoidf_(float x) { return 1.0f / (1.0f + __expf(-x)); }
__device__ __forceinline__ float eluf_(float x)     { return x > 0.f ? x : (__expf(x) - 1.f); }

// A fragment: 16x32 f16 tile from LDS, row-major [m][ldk] (K contiguous).
// Per ISA layout: lanes 0-15 hold M=lane, K={kb..kb+7, kb+16..kb+23};
//                 lanes 16-31 hold M=lane-16, K={kb+8..kb+15, kb+24..kb+31}.
__device__ __forceinline__ v16h load_frag_A(const half_t* base, int ldk, int kbase) {
    const int lane = threadIdx.x & 31;
    const int sub  = lane >> 4;
    const int m    = lane & 15;
    const half_t* p = base + m * ldk + kbase + sub * 8;
    v8h lo = *(const v8h*)(p);
    v8h hi = *(const v8h*)(p + 16);
    return __builtin_shufflevector(lo, hi, 0,1,2,3,4,5,6,7,8,9,10,11,12,13,14,15);
}

// B fragment: pre-swizzled fragment-major weights; each lane reads its 16
// contiguous halves (32 B) for tile `tile_idx`.
__device__ __forceinline__ v16h load_frag_B(const half_t* frag, int tile_idx) {
    const int lane = threadIdx.x & 31;
    const half_t* p = frag + ((size_t)tile_idx * 32 + lane) * 16;
    v8h lo = *(const v8h*)(p);
    v8h hi = *(const v8h*)(p + 8);
    return __builtin_shufflevector(lo, hi, 0,1,2,3,4,5,6,7,8,9,10,11,12,13,14,15);
}

__device__ __forceinline__ v8f wmma_f16(v16h a, v16h b, v8f c) {
    return __builtin_amdgcn_wmma_f32_16x16x32_f16(false, a, false, b, (short)0, c, false, false);
}

// ---------------------------------------------------------------------------
// Weight pre-swizzle: W[M,K] f32 row-major -> fragment-major f16, zero-padded.
// Fragment (nt,kt) holds B-tile for out-cols nt*16.., K-slice kt*32..
// Lane l<16 -> N=nt*16+l; half r -> K = kt*32 + r + (r>=8?8:0) + (l>=16?8:0).
// ---------------------------------------------------------------------------
__global__ void prep_frag_kernel(const float* __restrict__ W, int M, int K, int nkt,
                                 half_t* __restrict__ dst, int total) {
    int idx = blockIdx.x * blockDim.x + threadIdx.x;
    if (idx >= total) return;
    int fi     = idx >> 9;      // fragment index
    int within = idx & 511;
    int lane   = within >> 4;
    int r      = within & 15;
    int nt = fi / nkt;
    int kt = fi - nt * nkt;
    int sub  = lane >> 4;
    int n    = nt * 16 + (lane & 15);
    int koff = r + ((r >= 8) ? 8 : 0) + (sub ? 8 : 0);
    int k    = kt * 32 + koff;
    float v = (n < M && k < K) ? W[(size_t)n * K + k] : 0.f;
    dst[idx] = (half_t)v;
}

__global__ void prep_bsum_kernel(const float* __restrict__ a, const float* __restrict__ b,
                                 float* __restrict__ out, int n) {
    int i = blockIdx.x * blockDim.x + threadIdx.x;
    if (i < n) out[i] = a[i] + b[i];
}

// ---------------------------------------------------------------------------
// Generic WMMA LSTM: 16 batch rows / block, h/c resident in LDS, loops T steps.
// gates = x_t @ Wihᵀ + h @ Whhᵀ + (bih+bhh); PyTorch gate order i,f,g,o.
// lengths (optional): freeze h/c at t >= length[row].  reverse: iterate T-1..0.
// ---------------------------------------------------------------------------
#define LSTM_KI_MAX 192
#define LSTM_KH_MAX 192
#define LSTM_G_MAX  720
#define LSTM_H_MAX  180

__global__ __launch_bounds__(128)
void lstm_wmma_kernel(const float* __restrict__ x, int T, int I, int H,
                      int x_row_stride, int x_t_stride,
                      const half_t* __restrict__ wihF, const half_t* __restrict__ whhF,
                      const float* __restrict__ bsum, const int* __restrict__ lengths,
                      int reverse, float* __restrict__ h_out) {
    __shared__ __align__(16) half_t sh_x[16 * LSTM_KI_MAX];
    __shared__ __align__(16) half_t sh_h[16 * LSTM_KH_MAX];
    __shared__ half_t sh_g[16 * LSTM_G_MAX];
    __shared__ float  sh_c[16 * LSTM_H_MAX];

    const int tid  = threadIdx.x;
    const int wave = tid >> 5;
    const int lane = tid & 31;
    const int KIp  = (I + 31) & ~31;
    const int KHp  = (H + 31) & ~31;
    const int G    = 4 * H;
    const int NT   = G >> 4;
    const int nkt_i = KIp >> 5;
    const int nkt_h = KHp >> 5;
    const int row0 = blockIdx.x * 16;

    for (int idx = tid; idx < 16 * KHp; idx += blockDim.x) sh_h[idx] = (half_t)0.f;
    for (int idx = tid; idx < 16 * H;   idx += blockDim.x) sh_c[idx] = 0.f;
    __syncthreads();

    for (int t = 0; t < T; t++) {
        const int tt = reverse ? (T - 1 - t) : t;
        // stage x_t as f16, zero-padded to KIp
        for (int idx = tid; idx < 16 * KIp; idx += blockDim.x) {
            int m = idx / KIp, k = idx - m * KIp;
            float v = (k < I) ? x[(size_t)(row0 + m) * x_row_stride + (size_t)tt * x_t_stride + k] : 0.f;
            sh_x[m * KIp + k] = (half_t)v;
        }
        __syncthreads();

        // gates = x_t @ Wihᵀ + h @ Whhᵀ + bsum   (WMMA, 4 waves stride N-tiles)
        for (int nt = wave; nt < NT; nt += 4) {
            v8f acc = {0.f,0.f,0.f,0.f,0.f,0.f,0.f,0.f};
            for (int kt = 0; kt < nkt_i; kt++) {
                v16h a = load_frag_A(sh_x, KIp, kt * 32);
                v16h b = load_frag_B(wihF, nt * nkt_i + kt);
                acc = wmma_f16(a, b, acc);
            }
            for (int kt = 0; kt < nkt_h; kt++) {
                v16h a = load_frag_A(sh_h, KHp, kt * 32);
                v16h b = load_frag_B(whhF, nt * nkt_h + kt);
                acc = wmma_f16(a, b, acc);
            }
            const int sub = lane >> 4;
            const int col = nt * 16 + (lane & 15);
            const float bv = bsum[col];
            #pragma unroll
            for (int r = 0; r < 8; r++) {
                int m = r + sub * 8;                    // C/D layout: VGPR r -> M=r(+8)
                sh_g[m * G + col] = (half_t)(acc[r] + bv);
            }
        }
        __syncthreads();

        // elementwise gate update
        for (int idx = tid; idx < 16 * H; idx += blockDim.x) {
            int m = idx / H, j = idx - m * H;
            bool upd = true;
            if (lengths) upd = (t < lengths[row0 + m]);
            if (upd) {
                float gi = sigmoidf_((float)sh_g[m * G + j]);
                float gf = sigmoidf_((float)sh_g[m * G + H + j]);
                float gg = tanhf((float)sh_g[m * G + 2 * H + j]);
                float go = sigmoidf_((float)sh_g[m * G + 3 * H + j]);
                float c  = gf * sh_c[idx] + gi * gg;
                sh_c[idx] = c;
                sh_h[m * KHp + j] = (half_t)(go * tanhf(c));
            }
        }
        __syncthreads();
    }

    for (int idx = tid; idx < 16 * H; idx += blockDim.x) {
        int m = idx / H, j = idx - m * H;
        h_out[(size_t)(row0 + m) * H + j] = (float)sh_h[m * KHp + j];
    }
}

// ---------------------------------------------------------------------------
// Generic WMMA GEMM + bias + activation: out[N,M] = act(A[N,K] @ W[M,K]ᵀ + b)
// act: 0 = none, 1 = ELU, 2 = ReLU.  16 rows / block.
// ---------------------------------------------------------------------------
#define GEMM_KP_MAX 1088

__global__ __launch_bounds__(128)
void gemm_act_kernel(const float* __restrict__ A, int K, int M, int lda,
                     const half_t* __restrict__ wF, const float* __restrict__ bias,
                     float* __restrict__ out, int ldo, int act) {
    __shared__ __align__(16) half_t sh_A[16 * GEMM_KP_MAX];
    const int tid  = threadIdx.x;
    const int wave = tid >> 5;
    const int lane = tid & 31;
    const int KP   = (K + 31) & ~31;
    const int nkt  = KP >> 5;
    const int NT   = (M + 15) >> 4;
    const int row0 = blockIdx.x * 16;

    for (int idx = tid; idx < 16 * KP; idx += blockDim.x) {
        int m = idx / KP, k = idx - m * KP;
        sh_A[m * KP + k] = (half_t)((k < K) ? A[(size_t)(row0 + m) * lda + k] : 0.f);
    }
    __syncthreads();

    for (int nt = wave; nt < NT; nt += 4) {
        v8f acc = {0.f,0.f,0.f,0.f,0.f,0.f,0.f,0.f};
        for (int kt = 0; kt < nkt; kt++) {
            v16h a = load_frag_A(sh_A, KP, kt * 32);
            v16h b = load_frag_B(wF, nt * nkt + kt);
            acc = wmma_f16(a, b, acc);
        }
        const int sub = lane >> 4;
        const int col = nt * 16 + (lane & 15);
        if (col < M) {
            const float bv = bias[col];
            #pragma unroll
            for (int r = 0; r < 8; r++) {
                float v = acc[r] + bv;
                if (act == 1)      v = eluf_(v);
                else if (act == 2) v = fmaxf(v, 0.f);
                out[(size_t)(row0 + r + sub * 8) * ldo + col] = v;
            }
        }
    }
}

// ---------------------------------------------------------------------------
// Small helper kernels (concat / encode / predict)
// ---------------------------------------------------------------------------
__global__ void enc_kernel(const float* __restrict__ cv, const float* __restrict__ W,
                           const float* __restrict__ b, float* __restrict__ out, int rows) {
    int idx = blockIdx.x * blockDim.x + threadIdx.x;
    if (idx >= rows * 34) return;
    int r = idx / 34, o = idx - r * 34;
    const float* xr = cv + (size_t)r * 34;
    const float* wr = W + o * 34;
    float s = b[o];
    #pragma unroll
    for (int k = 0; k < 34; k++) s += xr[k] * wr[k];
    out[idx] = s;
}

__global__ void concat_comp_kernel(const float* __restrict__ cfp, const float* __restrict__ cefh,
                                   const float* __restrict__ cebh, const float* __restrict__ ctp,
                                   const float* __restrict__ exprh, float* __restrict__ out) {
    int idx = blockIdx.x * blockDim.x + threadIdx.x;
    const int total = 4096 * 1076;
    if (idx >= total) return;
    int n = idx / 1076, c = idx - n * 1076;
    float v;
    if      (c < 600) v = cfp[(size_t)n * 600 + c];
    else if (c < 700) v = cefh[(size_t)n * 100 + (c - 600)];
    else if (c < 800) v = cebh[(size_t)n * 100 + (c - 700)];
    else if (c < 976) v = ctp[(size_t)n * 176 + (c - 800)];
    else              v = exprh[(size_t)n * 100 + (c - 976)];
    out[idx] = v;
}

// concat [nodes_h(180) | comps_h(180) | loop(20)] -> [128,380]; ld*==0 broadcasts
__global__ void assemble_node_kernel(const float* __restrict__ nh, int ldn,
                                     const float* __restrict__ ch, int ldc,
                                     const float* __restrict__ lp, int ldl,
                                     float* __restrict__ out) {
    int idx = blockIdx.x * blockDim.x + threadIdx.x;
    if (idx >= 128 * 380) return;
    int n = idx / 380, c = idx - n * 380;
    float v;
    if      (c < 180) v = nh[(size_t)n * ldn + c];
    else if (c < 360) v = ch[(size_t)n * ldc + (c - 180)];
    else              v = lp[(size_t)n * ldl + (c - 360)];
    out[idx] = v;
}

__global__ void seq2_kernel(const float* __restrict__ a, const float* __restrict__ b,
                            float* __restrict__ out) {
    int idx = blockIdx.x * blockDim.x + threadIdx.x;
    if (idx >= 128 * 360) return;
    int n = idx / 360, j = idx - n * 360;
    out[idx] = (j < 180) ? a[(size_t)n * 180 + j] : b[(size_t)n * 180 + (j - 180)];
}

__global__ void pred_kernel(const float* __restrict__ x, const float* __restrict__ pw,
                            const float* __restrict__ pb, float* __restrict__ out) {
    int n = threadIdx.x;   // 128 threads, 1 block
    float s = pb[0];
    for (int j = 0; j < 180; j++) s += x[(size_t)n * 180 + j] * pw[j];
    out[n] = fmaxf(s, 0.f);
}

// ---------------------------------------------------------------------------
// Host orchestration
// ---------------------------------------------------------------------------
extern "C" void kernel_launch(void* const* d_in, const int* in_sizes, int n_in,
                              void* d_out, int out_size, void* d_ws, size_t ws_size,
                              hipStream_t stream) {
    (void)in_sizes; (void)n_in; (void)out_size; (void)ws_size;

    // Inputs in setup_inputs() insertion order, params depth-first insertion order.
    const float* cfp    = (const float*)d_in[0];   // [128,32,600]
    const float* cv     = (const float*)d_in[1];   // [4096,20,34]
    const float* ctp    = (const float*)d_in[2];   // [128,32,176]
    const float* lt     = (const float*)d_in[3];   // [128,40,20]
    const float* fcet   = (const float*)d_in[4];   // [128,32,64,11]
    const int*   exlen  = (const int*)  d_in[5];   // [4096]
    const float* enc_W  = (const float*)d_in[6];
    const float* enc_b  = (const float*)d_in[7];
    const float* eW_ih  = (const float*)d_in[8];   // expr [400,11]
    const float* eW_hh  = (const float*)d_in[9];   // [400,100]
    const float* eb_ih  = (const float*)d_in[10];
    const float* eb_hh  = (const float*)d_in[11];
    const float* fW_ih  = (const float*)d_in[12];  // cef [400,34]
    const float* fW_hh  = (const float*)d_in[13];
    const float* fb_ih  = (const float*)d_in[14];
    const float* fb_hh  = (const float*)d_in[15];
    const float* bW_ih  = (const float*)d_in[16];  // ceb
    const float* bW_hh  = (const float*)d_in[17];
    const float* bb_ih  = (const float*)d_in[18];
    const float* bb_hh  = (const float*)d_in[19];
    const float* cW_ih  = (const float*)d_in[20];  // comps [720,180]
    const float* cW_hh  = (const float*)d_in[21];
    const float* cb_ih  = (const float*)d_in[22];
    const float* cb_hh  = (const float*)d_in[23];
    const float* nW_ih  = (const float*)d_in[24];  // nodes
    const float* nW_hh  = (const float*)d_in[25];
    const float* nb_ih  = (const float*)d_in[26];
    const float* nb_hh  = (const float*)d_in[27];
    const float* compW0 = (const float*)d_in[28];  // [600,1076]
    const float* compW1 = (const float*)d_in[29];  // [350,600]
    const float* compW2 = (const float*)d_in[30];  // [200,350]
    const float* compW3 = (const float*)d_in[31];  // [180,200]
    const float* compb0 = (const float*)d_in[32];
    const float* compb1 = (const float*)d_in[33];
    const float* compb2 = (const float*)d_in[34];
    const float* compb3 = (const float*)d_in[35];
    const float* ccW0   = (const float*)d_in[36];  // [200,380]
    const float* ccW1   = (const float*)d_in[37];  // [180,200]
    const float* ccb0   = (const float*)d_in[38];
    const float* ccb1   = (const float*)d_in[39];
    const float* regW0  = (const float*)d_in[40];  // [200,180]
    const float* regW1  = (const float*)d_in[41];  // [180,200]
    const float* regb0  = (const float*)d_in[42];
    const float* regb1  = (const float*)d_in[43];
    const float* predW  = (const float*)d_in[44];  // [1,180]
    const float* predb  = (const float*)d_in[45];  // [1]
    const float* no_comps = (const float*)d_in[46]; // [1,180]
    const float* no_nodes = (const float*)d_in[47]; // [1,180]
    const float* root_it  = (const float*)d_in[48]; // [1,20]

    char* ws = (char*)d_ws;
    size_t off = 0;
    auto alloc = [&](size_t bytes) -> size_t {
        size_t o = off;
        off += (bytes + 255) & ~(size_t)255;
        return o;
    };
    auto fbuf = [&](size_t elems) -> float* { return (float*)(ws + alloc(elems * 4)); };

    // --- weight swizzle (fragment-major f16) ---
    auto frag = [&](const float* W, int M, int K) -> half_t* {
        int ntiles = (M + 15) / 16, nkt = (K + 31) / 32;
        int total = ntiles * nkt * 512;
        half_t* dst = (half_t*)(ws + alloc((size_t)total * 2));
        prep_frag_kernel<<<(total + 255) / 256, 256, 0, stream>>>(W, M, K, nkt, dst, total);
        return dst;
    };
    auto bsum = [&](const float* a, const float* b, int n) -> float* {
        float* dst = fbuf(n);
        prep_bsum_kernel<<<(n + 255) / 256, 256, 0, stream>>>(a, b, dst, n);
        return dst;
    };

    half_t* F_e_ih = frag(eW_ih, 400, 11);   half_t* F_e_hh = frag(eW_hh, 400, 100);
    half_t* F_f_ih = frag(fW_ih, 400, 34);   half_t* F_f_hh = frag(fW_hh, 400, 100);
    half_t* F_b_ih = frag(bW_ih, 400, 34);   half_t* F_b_hh = frag(bW_hh, 400, 100);
    half_t* F_c_ih = frag(cW_ih, 720, 180);  half_t* F_c_hh = frag(cW_hh, 720, 180);
    half_t* F_n_ih = frag(nW_ih, 720, 180);  half_t* F_n_hh = frag(nW_hh, 720, 180);
    half_t* F_cw0  = frag(compW0, 600, 1076);
    half_t* F_cw1  = frag(compW1, 350, 600);
    half_t* F_cw2  = frag(compW2, 200, 350);
    half_t* F_cw3  = frag(compW3, 180, 200);
    half_t* F_cc0  = frag(ccW0, 200, 380);
    half_t* F_cc1  = frag(ccW1, 180, 200);
    half_t* F_r0   = frag(regW0, 200, 180);
    half_t* F_r1   = frag(regW1, 180, 200);
    float* BS_e = bsum(eb_ih, eb_hh, 400);
    float* BS_f = bsum(fb_ih, fb_hh, 400);
    float* BS_b = bsum(bb_ih, bb_hh, 400);
    float* BS_c = bsum(cb_ih, cb_hh, 720);
    float* BS_n = bsum(nb_ih, nb_hh, 720);

    // --- activation buffers ---
    float* vecb  = fbuf((size_t)4096 * 20 * 34);
    float* exprh = fbuf((size_t)4096 * 100);
    float* cefh  = fbuf((size_t)4096 * 100);
    float* cebh  = fbuf((size_t)4096 * 100);
    float* cin   = fbuf((size_t)4096 * 1076);
    float* m1    = fbuf((size_t)4096 * 600);
    float* m2    = fbuf((size_t)4096 * 350);
    float* m3    = fbuf((size_t)4096 * 200);
    float* cemb  = fbuf((size_t)4096 * 180);   // [128,32,180]
    float* cl0   = fbuf(128 * 180);            // comps LSTM over idx 0..8
    float* cl1   = fbuf(128 * 180);            // idx 8..16
    float* cl2   = fbuf(128 * 180);            // idx 16..24
    float* hR0a  = fbuf(128 * 180);
    float* hR0b  = fbuf(128 * 180);
    float* hR0   = fbuf(128 * 180);
    float* hR1   = fbuf(128 * 180);
    float* nhR0  = fbuf(128 * 180);
    float* nhRt  = fbuf(128 * 180);
    float* pe    = fbuf(128 * 180);
    float* sq    = fbuf(128 * 2 * 180);
    float* catb  = fbuf(128 * 380);
    float* t200  = fbuf(128 * 200);
    float* t180  = fbuf(128 * 180);

    auto lstm = [&](const float* x, int N, int T, int I, int H, int rs, int ts,
                    half_t* wih, half_t* whh, float* bs, const int* len, int rev, float* hout) {
        lstm_wmma_kernel<<<N / 16, 128, 0, stream>>>(x, T, I, H, rs, ts, wih, whh, bs, len, rev, hout);
    };
    auto gemm = [&](const float* A, int N, int K, int M, int lda,
                    half_t* wF, const float* bias, float* out, int act) {
        gemm_act_kernel<<<N / 16, 128, 0, stream>>>(A, K, M, lda, wF, bias, out, M, act);
    };
    auto cat = [&](const float* nh, int ldn, const float* ch, int ldc,
                   const float* lp, int ldl, float* out) {
        assemble_node_kernel<<<(128 * 380 + 255) / 256, 256, 0, stream>>>(nh, ldn, ch, ldc, lp, ldl, out);
    };
    auto concat_mlp = [&](float* in, float* out) {   // 380->200->180, ELU both
        gemm(in, 128, 380, 200, 380, F_cc0, ccb0, t200, 1);
        gemm(t200, 128, 200, 180, 200, F_cc1, ccb1, out, 1);
    };

    // 1. encode vectors: vec = cv @ enc_Wᵀ + enc_b
    {
        int total = 81920 * 34;
        enc_kernel<<<(total + 255) / 256, 256, 0, stream>>>(cv, enc_W, enc_b, vecb, 81920);
    }

    // 2. the three big batch LSTMs (N=4096)
    lstm(fcet, 4096, 64, 11, 100, 64 * 11, 11, F_e_ih, F_e_hh, BS_e, exlen, 0, exprh);
    lstm(vecb, 4096, 20, 34, 100, 20 * 34, 34, F_f_ih, F_f_hh, BS_f, nullptr, 0, cefh);
    lstm(vecb, 4096, 20, 34, 100, 20 * 34, 34, F_b_ih, F_b_hh, BS_b, nullptr, 1, cebh);

    // 3. comp embedding MLP: concat -> 600 -> 350 -> 200 -> 180 (ELU)
    concat_comp_kernel<<<(4096 * 1076 + 255) / 256, 256, 0, stream>>>(cfp, cefh, cebh, ctp, exprh, cin);
    gemm(cin, 4096, 1076, 600, 1076, F_cw0, compb0, m1, 1);
    gemm(m1,  4096, 600,  350, 600,  F_cw1, compb1, m2, 1);
    gemm(m2,  4096, 350,  200, 350,  F_cw2, compb2, m3, 1);
    gemm(m3,  4096, 200,  180, 200,  F_cw3, compb3, cemb, 1);

    // 4. tree: comps LSTMs over comp index segments (strided view of cemb [128,32,180])
    lstm(cemb + 0 * 180,  128, 8, 180, 180, 32 * 180, 180, F_c_ih, F_c_hh, BS_c, nullptr, 0, cl0);
    lstm(cemb + 8 * 180,  128, 8, 180, 180, 32 * 180, 180, F_c_ih, F_c_hh, BS_c, nullptr, 0, cl1);
    lstm(cemb + 16 * 180, 128, 8, 180, 180, 32 * 180, 180, F_c_ih, F_c_hh, BS_c, nullptr, 0, cl2);

    // R0a: no children, comps 8..16, loop 1
    cat(no_nodes, 0, cl1, 180, lt + 1 * 20, 40 * 20, catb);  concat_mlp(catb, hR0a);
    // R0b: no children, no comps, loop 2
    cat(no_nodes, 0, no_comps, 0, lt + 2 * 20, 40 * 20, catb);  concat_mlp(catb, hR0b);
    // R0: nodes LSTM over [hR0a, hR0b], comps 0..8, loop 0
    seq2_kernel<<<(128 * 360 + 255) / 256, 256, 0, stream>>>(hR0a, hR0b, sq);
    lstm(sq, 128, 2, 180, 180, 2 * 180, 180, F_n_ih, F_n_hh, BS_n, nullptr, 0, nhR0);
    cat(nhR0, 180, cl0, 180, lt + 0 * 20, 40 * 20, catb);  concat_mlp(catb, hR0);
    // R1: no children, comps 16..24, loop 3
    cat(no_nodes, 0, cl2, 180, lt + 3 * 20, 40 * 20, catb);  concat_mlp(catb, hR1);
    // root: nodes LSTM over [hR0, hR1], no comps, root_iterator loop
    seq2_kernel<<<(128 * 360 + 255) / 256, 256, 0, stream>>>(hR0, hR1, sq);
    lstm(sq, 128, 2, 180, 180, 2 * 180, 180, F_n_ih, F_n_hh, BS_n, nullptr, 0, nhRt);
    cat(nhRt, 180, no_comps, 0, root_it, 0, catb);  concat_mlp(catb, pe);

    // 5. regression head: 180 -> 200 -> 180 (ELU), then linear + ReLU
    gemm(pe,   128, 180, 200, 180, F_r0, regb0, t200, 1);
    gemm(t200, 128, 200, 180, 200, F_r1, regb1, t180, 1);
    pred_kernel<<<1, 128, 0, stream>>>(t180, predW, predb, (float*)d_out);
}
